// Compressor_87041807221472
// MI455X (gfx1250) — matrix operand order repaired and checked by reading, not compiled, not run on hardware
//
#include <hip/hip_runtime.h>
#include <hip/hip_bf16.h>
#include <stdint.h>

// Problem constants (from reference): B=16, T=2048, D=1024
#define BSZ 16
#define TT  2048
#define DM  1024
#define BT  (BSZ*TT)

// ---- plain trivial vector types (safe inside unions) ----
typedef __attribute__((ext_vector_type(16))) __bf16        v16bf;
typedef __attribute__((ext_vector_type(8)))  float         v8f;
typedef __attribute__((ext_vector_type(4)))  unsigned int  u32x4;
typedef __attribute__((ext_vector_type(2)))  unsigned int  u32x2;
typedef __attribute__((ext_vector_type(4)))  float         f32x4;
typedef __attribute__((ext_vector_type(4)))  int           i32v4;
typedef __attribute__((ext_vector_type(2)))  int           i32v2;

// gfx1250 async global->LDS path (ASYNCcnt), guarded so either toolchain compiles
#if defined(__has_builtin)
#if __has_builtin(__builtin_amdgcn_global_load_async_to_lds_b128)
#define HAVE_ASYNC_B128 1
#endif
#if __has_builtin(__builtin_amdgcn_global_load_async_to_lds_b64)
#define HAVE_ASYNC_B64 1
#endif
#endif

// exact pointee types per hipcc diagnostic: v4i / v2i with AS1 (global), AS3 (LDS)
typedef __attribute__((address_space(1))) i32v4 as1_i32v4;
typedef __attribute__((address_space(3))) i32v4 as3_i32v4;
typedef __attribute__((address_space(1))) i32v2 as1_i32v2;
typedef __attribute__((address_space(3))) i32v2 as3_i32v2;

__device__ __forceinline__ void wait_async0() {
#if defined(__has_builtin)
#if __has_builtin(__builtin_amdgcn_s_wait_asynccnt)
  __builtin_amdgcn_s_wait_asynccnt(0);
  return;
#else
  asm volatile("s_wait_asynccnt 0x0" ::: "memory");
  return;
#endif
#endif
}

__device__ __forceinline__ unsigned short f2bf(float f) {
  unsigned int u = __float_as_uint(f);
  u += 0x7FFFu + ((u >> 16) & 1u);          // round-to-nearest-even
  return (unsigned short)(u >> 16);
}
__device__ __forceinline__ float bf2f(unsigned short h) {
  return __uint_as_float(((unsigned int)h) << 16);
}
__device__ __forceinline__ unsigned int pack2(float lo, float hi) {
  return (unsigned int)f2bf(lo) | ((unsigned int)f2bf(hi) << 16);
}

// inclusive Hillis-Steele scan over 256 ints in LDS; sh[255] = total afterwards
__device__ __forceinline__ int block_scan_incl(int* sh, int tid, int v) {
  sh[tid] = v; __syncthreads();
  #pragma unroll
  for (int o = 1; o < 256; o <<= 1) {
    int t = (tid >= o) ? sh[tid - o] : 0;
    __syncthreads();
    sh[tid] += t;
    __syncthreads();
  }
  return sh[tid];
}

// ---------------------------------------------------------------------------
// K1: per-batch-row segment scan (unique_consecutive structure)
// ---------------------------------------------------------------------------
__global__ __launch_bounds__(256) void seg_scan_kernel(
    const int* __restrict__ preds,
    int* __restrict__ seg_start, int* __restrict__ seg_pred,
    int* __restrict__ nseg)
{
  __shared__ int sh[256];
  const int b = blockIdx.x, tid = threadIdx.x;
  const int* p = preds + b * TT;
  int bnd[8]; int c = 0;
  #pragma unroll
  for (int i = 0; i < 8; ++i) {
    int t = tid * 8 + i;
    int bb = (t == 0) || (p[t] != p[t - 1]);
    bnd[i] = bb; c += bb;
  }
  int incl  = block_scan_incl(sh, tid, c);
  int total = sh[255];
  int run   = incl - c;                       // exclusive prefix
  #pragma unroll
  for (int i = 0; i < 8; ++i) {
    int t = tid * 8 + i;
    run += bnd[i];
    int s = run - 1;                          // seg = cumsum(boundary)-1
    if (bnd[i]) { seg_start[b * TT + s] = t; seg_pred[b * TT + s] = p[t]; }
  }
  if (tid == 0) nseg[b] = total;
}

// ---------------------------------------------------------------------------
// K1b: segment lengths + left-packed positions (blank segments dropped)
// ---------------------------------------------------------------------------
__global__ __launch_bounds__(256) void seg_pack_kernel(
    const int* __restrict__ seg_start, const int* __restrict__ seg_pred,
    const int* __restrict__ nseg,
    int* __restrict__ seg_len, int* __restrict__ seg_pos,
    int* __restrict__ lens)
{
  __shared__ int sh[256];
  const int b = blockIdx.x, tid = threadIdx.x;
  const int ns = nseg[b];
  int keep[8]; int c = 0;
  #pragma unroll
  for (int i = 0; i < 8; ++i) {
    int s = tid * 8 + i, base = b * TT + s;
    int ln = 0;
    if (s < ns) {
      int st = seg_start[base];
      int en = (s + 1 < ns) ? seg_start[base + 1] : TT;
      ln = en - st;
    }
    seg_len[base] = ln;
    int kp = (s < ns) && (seg_pred[base] != 0);   // BLANK == 0
    keep[i] = kp; c += kp;
  }
  int incl  = block_scan_incl(sh, tid, c);
  int total = sh[255];
  int run   = incl - c;
  #pragma unroll
  for (int i = 0; i < 8; ++i) {
    int s = tid * 8 + i, base = b * TT + s;
    int pos = run; run += keep[i];
    seg_pos[base] = keep[i] ? pos : -1;
  }
  if (tid == 0) lens[b] = total;
}

// ---------------------------------------------------------------------------
// K-W: one-time fp32 -> bf16 weight conversion + tile transpose.
// Wt layout: [m 0..3][n_tile 0..7][k_step 0..31][col 0..127][k 0..31] bf16:
// each (m, n_tile, k_step) tile is a contiguous 8 KB N-major block that the
// GEMMs stage into LDS with pure async b128 copies (no VALU).
// ---------------------------------------------------------------------------
__global__ __launch_bounds__(256) void wconv_kernel(
    const float* __restrict__ W0, const float* __restrict__ W1,
    const float* __restrict__ W2, const float* __restrict__ W3,
    unsigned int* __restrict__ Wt32)
{
  const float* Ws[4] = { W0, W1, W2, W3 };
  const size_t total  = (size_t)4 * 1024 * 512;       // u32 elements (k-pairs)
  const size_t stride = (size_t)gridDim.x * blockDim.x;
  for (size_t e = (size_t)blockIdx.x * blockDim.x + threadIdx.x;
       e < total; e += stride) {
    int m   = (int)(e >> 19);                 // 512K u32 per matrix
    int rem = (int)(e & ((1u << 19) - 1));
    int kp  = rem >> 10;                      // 0..511 (pair of k rows)
    int n   = rem & 1023;
    int k   = kp * 2;
    const float* W = Ws[m];
    float f0 = W[(size_t)k * DM + n];         // coalesced over n
    float f1 = W[(size_t)(k + 1) * DM + n];
    int nit = n >> 7, col = n & 127, ks = k >> 5, kl = k & 31;
    size_t dst = ((size_t)((m * 8 + nit) * 32 + ks) * 4096 + col * 32 + kl) >> 1;
    Wt32[dst] = pack2(f0, f1);
  }
}

// ---------------------------------------------------------------------------
// K-X: one-time fp32 -> bf16 conversion of x (row-major layout preserved).
// Converts each element once instead of 8x inside the GEMM N-loop, and
// halves the GEMM's x DRAM traffic.
// ---------------------------------------------------------------------------
__global__ __launch_bounds__(256) void xconv_kernel(
    const float* __restrict__ x, unsigned int* __restrict__ xb32)
{
  const size_t total  = (size_t)BT * DM / 2;          // u32 elements
  const size_t stride = (size_t)gridDim.x * blockDim.x;
  for (size_t e = (size_t)blockIdx.x * blockDim.x + threadIdx.x;
       e < total; e += stride) {
    float f0 = x[2 * e];
    float f1 = x[2 * e + 1];
    xb32[e] = pack2(f0, f1);
  }
}

// Stage one pre-transposed 8 KB bf16 weight tile into LDS:
// 2 async b128 copies per thread (fallback: u32x4 copies).
__device__ __forceinline__ void stage_wt_tile(
    unsigned short* lds, const unsigned short* __restrict__ Wt,
    int m, int nit, int kk, int tid)
{
  const unsigned short* tsrc =
      Wt + (size_t)((m * 8 + nit) * 32 + kk) * 4096;
  const int o = tid * 8;                       // 16B chunk (ushort units)
#ifdef HAVE_ASYNC_B128
  __builtin_amdgcn_global_load_async_to_lds_b128(
      (as1_i32v4*)(tsrc + o), (as3_i32v4*)(lds + o), 0, 0);
  __builtin_amdgcn_global_load_async_to_lds_b128(
      (as1_i32v4*)(tsrc + 2048 + o), (as3_i32v4*)(lds + 2048 + o), 0, 0);
#else
  *(u32x4*)(lds + o)        = *(const u32x4*)(tsrc + o);
  *(u32x4*)(lds + 2048 + o) = *(const u32x4*)(tsrc + 2048 + o);
#endif
}

// Stage a 32row x 32k bf16 activation tile (row-major) into LDS:
// 1 async b64 copy per thread (fallback: u32x2 copy).
__device__ __forceinline__ void stage_a_tile(
    unsigned short* lds, const unsigned short* __restrict__ src_rows,
    int rowbase, int k0, int tid)
{
  const int idx = tid * 4;                     // 8B chunk (ushort units)
  const unsigned short* src =
      &src_rows[(size_t)(rowbase + (idx >> 5)) * DM + k0 + (idx & 31)];
#ifdef HAVE_ASYNC_B64
  __builtin_amdgcn_global_load_async_to_lds_b64(
      (as1_i32v2*)src, (as3_i32v2*)(lds + idx), 0, 0);
#else
  *(u32x2*)(lds + idx) = *(const u32x2*)src;
#endif
}

// A fragment: two raw b128 LDS reads in the ISA 16x32 bf16 layout.
__device__ __forceinline__ v16bf load_a_frag(const unsigned short* lds,
                                             int mrow0, int l4, int hi)
{
  union { v16bf v; u32x4 q[2]; } fr;
  const unsigned short* ap = &lds[(mrow0 + l4) * 32 + hi * 8];
  fr.q[0] = *(const u32x4*)(ap);        // K 0..7   (hi: 8..15)
  fr.q[1] = *(const u32x4*)(ap + 16);   // K 16..23 (hi: 24..31)
  return fr.v;
}

// ---------------------------------------------------------------------------
// K2: fused q/k/v GEMM.  Block tile = 32 rows x 128 cols, 8 waves (2M x 4N),
// each wave owns two 16x16 output subtiles per weight matrix.  A (x tile) is
// shared across the 3 weight matrices -> 3 WMMAs per k-step per subtile.
// All tiles (bf16 x + pre-transposed bf16 weights) staged with async copies:
// the inner loop is pure ASYNC-DMA -> ds_load_b128 -> v_wmma.
// Epilogue: scores += (q+bq)*(k+bk) via ds_add_f32; v+bv streamed as bf16.
// ---------------------------------------------------------------------------
__global__ __launch_bounds__(256) void qkv_score_kernel(
    const unsigned short* __restrict__ xb, const unsigned short* __restrict__ Wt,
    const float* __restrict__ bq, const float* __restrict__ bk,
    const float* __restrict__ bv,
    float* __restrict__ scores, unsigned short* __restrict__ vbuf)
{
  __shared__ __align__(16) unsigned short lds_a[32 * 32];       //  2 KB bf16
  __shared__ __align__(16) unsigned short lds_w[3 * 128 * 32];  // 24 KB bf16
  __shared__ float lds_score[32];

  const int tid  = threadIdx.x;
  const int lane = tid & 31, wid = tid >> 5;
  const int hi   = lane >> 4;           // half-wave select (ISA frag layout)
  const int l4   = lane & 15;
  const int mw   = wid & 1;             // 2 M-subtiles of 16 rows
  const int nwq  = wid >> 1;            // 4 N-groups of 32 cols
  const int mrow0   = mw * 16;
  const int rowbase = blockIdx.x * 32;

  if (tid < 32) lds_score[tid] = 0.0f;
  __syncthreads();

  const v8f zero8 = { 0, 0, 0, 0, 0, 0, 0, 0 };

  for (int nit = 0; nit < 8; ++nit) {           // 8 x 128 = 1024 output cols
    v8f acc[2][3];
    #pragma unroll
    for (int i = 0; i < 2; ++i)
      #pragma unroll
      for (int m = 0; m < 3; ++m) acc[i][m] = zero8;

    for (int kk = 0; kk < 32; ++kk) {           // K = 1024 in steps of 32
      const int k0 = kk * 32;
      stage_a_tile(lds_a, xb, rowbase, k0, tid);
      #pragma unroll
      for (int m = 0; m < 3; ++m)
        stage_wt_tile(&lds_w[m * 4096], Wt, m, nit, kk, tid);

      if (kk + 1 < 32)   // gfx1250 prefetch path (global_prefetch_b8)
        __builtin_prefetch(&xb[(size_t)(rowbase + (tid >> 3)) * DM + k0 + 32], 0, 1);
#if defined(HAVE_ASYNC_B64) || defined(HAVE_ASYNC_B128)
      wait_async0();
#endif
      __syncthreads();

      // ---- A fragment shared by q/k/v WMMAs ----
      v16bf av = load_a_frag(lds_a, mrow0, l4, hi);
      #pragma unroll
      for (int ns2 = 0; ns2 < 2; ++ns2) {
        const int ncl = nwq * 32 + ns2 * 16;
        #pragma unroll
        for (int m = 0; m < 3; ++m) {
          union { v16bf v; u32x4 q[2]; } bfr;
          const unsigned short* bp = &lds_w[m * 4096 + (ncl + l4) * 32 + hi * 8];
          bfr.q[0] = *(const u32x4*)(bp);       // K 0..7 (hi: 8..15)
          bfr.q[1] = *(const u32x4*)(bp + 16);  // K 16..23 (hi: 24..31)
          acc[ns2][m] = __builtin_amdgcn_wmma_f32_16x16x32_bf16(
              false, av, false, bfr.v, (short)0, acc[ns2][m], false, false);
        }
      }
      __syncthreads();                          // protect next staging
    }

    // ---- epilogue for this 128-col slab ----
    #pragma unroll
    for (int ns2 = 0; ns2 < 2; ++ns2) {
      const int gcol = nit * 128 + nwq * 32 + ns2 * 16 + l4;
      const float bqv = bq[gcol], bkv = bk[gcol], bvv = bv[gcol];
      #pragma unroll
      for (int r = 0; r < 8; ++r) {             // C frag: M = r + 8*hi, N = l4
        int lrow = mrow0 + r + 8 * hi;
        float qv = acc[ns2][0][r] + bqv;
        float kv = acc[ns2][1][r] + bkv;
        atomicAdd(&lds_score[lrow], qv * kv);   // ds_add_f32
        float vv = acc[ns2][2][r] + bvv;
        vbuf[(size_t)(rowbase + lrow) * DM + gcol] = f2bf(vv);
      }
    }
  }
  __syncthreads();
  if (tid < 32) scores[rowbase + tid] = lds_score[tid] * 0.03125f; // 1/sqrt(1024)
}

// ---------------------------------------------------------------------------
// K3: segment softmax-pooling.  One wave per segment; segments are contiguous
// token ranges, so no atomics: wave accumulates w*v over its tokens.
// ---------------------------------------------------------------------------
__global__ __launch_bounds__(256) void pool_kernel(
    const float* __restrict__ scores, const unsigned short* __restrict__ vbuf,
    const int* __restrict__ seg_start, const int* __restrict__ seg_len,
    const int* __restrict__ nseg, unsigned short* __restrict__ pooled)
{
  const int lane = threadIdx.x & 31, wid = threadIdx.x >> 5;
  const int g = blockIdx.x * 8 + wid;           // global segment slot
  const int b = g >> 11, s = g & (TT - 1);
  const unsigned int* v32 = (const unsigned int*)vbuf;
  unsigned int* p32 = (unsigned int*)pooled;
  const size_t rowoff32 = (size_t)g * (DM / 2);
  if (s >= nseg[b]) {                           // empty slot: zero the row
    #pragma unroll
    for (int j = 0; j < 16; ++j) p32[rowoff32 + j * 32 + lane] = 0u;
    return;
  }
  const int st = seg_start[g], ln = seg_len[g];
  const int tokbase = b * TT + st;
  float mx = -3.402823466e38f;
  for (int t = 0; t < ln; ++t) mx = fmaxf(mx, scores[tokbase + t]);
  float den = 0.f;
  for (int t = 0; t < ln; ++t) den += __expf(scores[tokbase + t] - mx);
  const float inv = 1.0f / den;

  float acc[32];
  #pragma unroll
  for (int j = 0; j < 32; ++j) acc[j] = 0.f;
  for (int t = 0; t < ln; ++t) {
    float w = __expf(scores[tokbase + t] - mx) * inv;
    const unsigned int* vp = v32 + (size_t)(tokbase + t) * (DM / 2);
    #pragma unroll
    for (int j = 0; j < 16; ++j) {
      unsigned int u = vp[j * 32 + lane];
      acc[2 * j]     += w * bf2f((unsigned short)(u & 0xFFFFu));
      acc[2 * j + 1] += w * bf2f((unsigned short)(u >> 16));
    }
  }
  #pragma unroll
  for (int j = 0; j < 16; ++j)
    p32[rowoff32 + j * 32 + lane] = pack2(acc[2 * j], acc[2 * j + 1]);
}

// ---------------------------------------------------------------------------
// K4: pooled @ Wo + bo, scattering rows to left-packed output positions.
// Same DMA-fed WMMA pipeline (Wo = matrix 3 of the pre-transposed weights).
// ---------------------------------------------------------------------------
__global__ __launch_bounds__(256) void out_gemm_kernel(
    const unsigned short* __restrict__ pooled,
    const unsigned short* __restrict__ Wt, const float* __restrict__ bo,
    const int* __restrict__ seg_pos, float* __restrict__ out)
{
  __shared__ __align__(16) unsigned short lds_a[32 * 32];    // 2 KB bf16
  __shared__ __align__(16) unsigned short lds_w[128 * 32];   // 8 KB bf16
  const int tid = threadIdx.x, lane = tid & 31, wid = tid >> 5;
  const int hi = lane >> 4, l4 = lane & 15;
  const int mw = wid & 1, nwq = wid >> 1;
  const int mrow0 = mw * 16;
  const int rowbase = blockIdx.x * 32;
  const v8f zero8 = { 0, 0, 0, 0, 0, 0, 0, 0 };

  for (int nit = 0; nit < 8; ++nit) {
    v8f acc[2]; acc[0] = zero8; acc[1] = zero8;
    for (int kk = 0; kk < 32; ++kk) {
      const int k0 = kk * 32;
      stage_a_tile(lds_a, pooled, rowbase, k0, tid);
      stage_wt_tile(lds_w, Wt, 3, nit, kk, tid);
#if defined(HAVE_ASYNC_B64) || defined(HAVE_ASYNC_B128)
      wait_async0();
#endif
      __syncthreads();

      v16bf av = load_a_frag(lds_a, mrow0, l4, hi);
      #pragma unroll
      for (int ns2 = 0; ns2 < 2; ++ns2) {
        union { v16bf v; u32x4 q[2]; } bfr;
        const unsigned short* bp = &lds_w[(nwq * 32 + ns2 * 16 + l4) * 32 + hi * 8];
        bfr.q[0] = *(const u32x4*)bp;
        bfr.q[1] = *(const u32x4*)(bp + 16);
        acc[ns2] = __builtin_amdgcn_wmma_f32_16x16x32_bf16(
            false, av, false, bfr.v, (short)0, acc[ns2], false, false);
      }
      __syncthreads();
    }
    #pragma unroll
    for (int ns2 = 0; ns2 < 2; ++ns2) {
      const int gcol = nit * 128 + nwq * 32 + ns2 * 16 + l4;
      const float bov = bo[gcol];
      #pragma unroll
      for (int r = 0; r < 8; ++r) {
        int lrow = mrow0 + r + 8 * hi;
        int grow = rowbase + lrow;
        int pos = seg_pos[grow];
        if (pos >= 0) {
          int bb = grow >> 11;
          out[((size_t)bb * TT + pos) * DM + gcol] = acc[ns2][r] + bov;
        }
      }
    }
  }
}

// ---------------------------------------------------------------------------
// K0 / K5: output init + finalize (mask / lens / packed preds)
// ---------------------------------------------------------------------------
__global__ void init_out_kernel(float* __restrict__ out, float* __restrict__ predsF)
{
  const size_t stride = (size_t)gridDim.x * blockDim.x;
  const size_t i0 = (size_t)blockIdx.x * blockDim.x + threadIdx.x;
  for (size_t k = i0; k < (size_t)BT * DM; k += stride) out[k] = 0.f;
  for (size_t k = i0; k < (size_t)BT; k += stride) predsF[k] = 0.f;
}

__global__ void finalize_kernel(
    const int* __restrict__ lens, const int* __restrict__ seg_pos,
    const int* __restrict__ seg_pred,
    float* __restrict__ maskF, float* __restrict__ lensF, float* __restrict__ predsF)
{
  const int b = blockIdx.x, tid = threadIdx.x;
  const int ln = lens[b];
  if (tid == 0) lensF[b] = (float)ln;
  for (int t = tid; t < TT; t += 256) maskF[b * TT + t] = (t >= ln) ? 1.0f : 0.0f;
  for (int s = tid; s < TT; s += 256) {
    int pos = seg_pos[b * TT + s];
    if (pos >= 0) predsF[b * TT + pos] = (float)seg_pred[b * TT + s];
  }
}

// ---------------------------------------------------------------------------
extern "C" void kernel_launch(void* const* d_in, const int* in_sizes, int n_in,
                              void* d_out, int out_size, void* d_ws, size_t ws_size,
                              hipStream_t stream)
{
  (void)in_sizes; (void)n_in; (void)out_size; (void)ws_size;
  const float* x     = (const float*)d_in[0];
  const int*   preds = (const int*)  d_in[1];
  const float* Wq = (const float*)d_in[2]; const float* bq = (const float*)d_in[3];
  const float* Wk = (const float*)d_in[4]; const float* bk = (const float*)d_in[5];
  const float* Wv = (const float*)d_in[6]; const float* bv = (const float*)d_in[7];
  const float* Wo = (const float*)d_in[8]; const float* bo = (const float*)d_in[9];

  // workspace layout (~201 MB): ints + scores + bf16 {x, v, pooled, Wt}
  char* w = (char*)d_ws;
  int* seg_start = (int*)w; w += (size_t)BT * 4;
  int* seg_pred  = (int*)w; w += (size_t)BT * 4;
  int* seg_len   = (int*)w; w += (size_t)BT * 4;
  int* seg_pos   = (int*)w; w += (size_t)BT * 4;
  int* nseg      = (int*)w; w += 256;
  int* lens      = (int*)w; w += 256;
  float* scores  = (float*)w; w += (size_t)BT * 4;
  unsigned short* vbuf   = (unsigned short*)w; w += (size_t)BT * DM * 2;
  unsigned short* pooled = (unsigned short*)w; w += (size_t)BT * DM * 2;
  unsigned short* Wt     = (unsigned short*)w; w += (size_t)4 * DM * DM * 2;
  unsigned short* xb     = (unsigned short*)w; w += (size_t)BT * DM * 2;

  // output layout: out [B,T,D] f32, then mask [B,T], lens [B], preds [B,T]
  float* out    = (float*)d_out;
  float* maskF  = out + (size_t)BT * DM;
  float* lensF  = maskF + BT;
  float* predsF = lensF + BSZ;

  init_out_kernel   <<<8192, 256, 0, stream>>>(out, predsF);
  wconv_kernel      <<<4096, 256, 0, stream>>>(Wq, Wk, Wv, Wo, (unsigned int*)Wt);
  xconv_kernel      <<<8192, 256, 0, stream>>>(x, (unsigned int*)xb);
  seg_scan_kernel   <<<BSZ,  256, 0, stream>>>(preds, seg_start, seg_pred, nseg);
  seg_pack_kernel   <<<BSZ,  256, 0, stream>>>(seg_start, seg_pred, nseg,
                                               seg_len, seg_pos, lens);
  qkv_score_kernel  <<<BT/32, 256, 0, stream>>>(xb, Wt, bq, bk, bv, scores, vbuf);
  pool_kernel       <<<BT/8,  256, 0, stream>>>(scores, vbuf, seg_start, seg_len,
                                                nseg, pooled);
  out_gemm_kernel   <<<BT/32, 256, 0, stream>>>(pooled, Wt, bo, seg_pos, out);
  finalize_kernel   <<<BSZ,   256, 0, stream>>>(lens, seg_pos, seg_pred,
                                                maskF, lensF, predsF);
}
